// BasicAttention_27874337751529
// MI455X (gfx1250) — compile-verified, matrix operand order checked by reading, not run on hardware
//
#include <hip/hip_runtime.h>
#include <hip/hip_bf16.h>

// Problem dims (from reference)
#define BB 16
#define CC 1024
#define QQ 128
#define HH 768

typedef __bf16 bf16_t;
typedef __attribute__((ext_vector_type(16))) __bf16 v16bf;
typedef __attribute__((ext_vector_type(8)))  __bf16 v8bf;
typedef __attribute__((ext_vector_type(8)))  float  v8f;

__device__ __forceinline__ v8f vzero8() {
    v8f z;
#pragma unroll
    for (int i = 0; i < 8; ++i) z[i] = 0.0f;
    return z;
}

// Load a 16x32 bf16 WMMA A/B fragment for the X @ Y^T pattern from a
// row-major [rows, ld] bf16 matrix. Per CDNA5 ISA 7.12.2 (16-bit A layout):
//   lane L holds row (row0 + L&15); element i maps to
//   k = k0 + 8*(L>>4) + (i<8 ? i : 16 + (i-8))
// i.e. two contiguous 8-element (16-byte) chunks.
__device__ __forceinline__ v16bf load_frag_bf16(const bf16_t* __restrict__ base,
                                                int ld, int row0, int k0, int lane) {
    const bf16_t* p = base + (size_t)(row0 + (lane & 15)) * (size_t)ld
                           + (size_t)(k0 + ((lane >> 4) << 3));
    v8bf lo = *(const v8bf*)(p);
    v8bf hi = *(const v8bf*)(p + 16);
    return __builtin_shufflevector(lo, hi, 0,1,2,3,4,5,6,7,8,9,10,11,12,13,14,15);
}

__device__ __forceinline__ v8f wmma_bf16(v16bf a, v16bf b, v8f c) {
    // (neg_a, A, neg_b, B, c_mod, C, reuse_a, reuse_b)
    return __builtin_amdgcn_wmma_f32_16x16x32_bf16(false, a, false, b, (short)0, c,
                                                   false, false);
}

// ---------------------------------------------------------------- cvt f32->bf16
__global__ void __launch_bounds__(256)
cvt_f32_bf16_kernel(const float* __restrict__ src, bf16_t* __restrict__ dst, int n) {
    int i = blockIdx.x * 256 + threadIdx.x;
    int stride = gridDim.x * 256;
    for (; i < n; i += stride) dst[i] = (bf16_t)src[i];
}

// ------------------------------------------------------- qry projection (GEMM)
// qry = query @ Wq^T + bq      rows = B*Q, K = N = H
// writes bf16 row-major qry and bf16 transposed qryT[b][p][q]
__global__ void __launch_bounds__(256)
proj_q_kernel(const bf16_t* __restrict__ Xbf, const bf16_t* __restrict__ Wbf,
              const float* __restrict__ bias,
              bf16_t* __restrict__ qry_bf, bf16_t* __restrict__ qryT_bf) {
    const int lane = threadIdx.x & 31;
    const int wave = threadIdx.x >> 5;
    const int half = lane >> 4;
    const int m0   = blockIdx.x * 128 + wave * 16;   // global row in [0, B*Q)

    for (int n0 = 0; n0 < HH; n0 += 16) {
        v8f acc = vzero8();
        for (int k0 = 0; k0 < HH; k0 += 32) {
            v16bf a = load_frag_bf16(Xbf, HH, m0, k0, lane);
            v16bf b = load_frag_bf16(Wbf, HH, n0, k0, lane);
            acc = wmma_bf16(a, b, acc);
        }
        const int n  = n0 + (lane & 15);
        const float bv = bias[n];
#pragma unroll
        for (int v = 0; v < 8; ++v) {
            const int r = m0 + half * 8 + v;          // global row
            const int bidx = r >> 7;                  // / QQ
            const int q    = r & (QQ - 1);
            const float y  = acc[v] + bv;
            qry_bf[(size_t)r * HH + n] = (bf16_t)y;
            qryT_bf[((size_t)bidx * HH + n) * QQ + q] = (bf16_t)y;
        }
    }
}

// ------------------------------------------------------- ctx projection (GEMM)
// ctx = context @ Wc^T + bc      rows = B*C
// writes f32 ctx into out[..., 0:H] and bf16 (ctx * w_att) for the sim GEMM
__global__ void __launch_bounds__(256)
proj_c_kernel(const bf16_t* __restrict__ Xbf, const bf16_t* __restrict__ Wbf,
              const float* __restrict__ bias, const float* __restrict__ w_att,
              float* __restrict__ out, bf16_t* __restrict__ ctxw_bf) {
    const int lane = threadIdx.x & 31;
    const int wave = threadIdx.x >> 5;
    const int half = lane >> 4;
    const int m0   = blockIdx.x * 128 + wave * 16;   // global row in [0, B*C)

    for (int n0 = 0; n0 < HH; n0 += 16) {
        v8f acc = vzero8();
        for (int k0 = 0; k0 < HH; k0 += 32) {
            v16bf a = load_frag_bf16(Xbf, HH, m0, k0, lane);
            v16bf b = load_frag_bf16(Wbf, HH, n0, k0, lane);
            acc = wmma_bf16(a, b, acc);
        }
        const int n   = n0 + (lane & 15);
        const float bv = bias[n];
        const float wv = w_att[n];
#pragma unroll
        for (int v = 0; v < 8; ++v) {
            const int r = m0 + half * 8 + v;          // global ctx row (b*C + c)
            const float y = acc[v] + bv;
            out[(size_t)r * (4 * HH) + n] = y;        // ctx slot
            ctxw_bf[(size_t)r * HH + n]   = (bf16_t)(y * wv);
        }
    }
}

// --------------------------------------- fused sim GEMM + softmax + a GEMM + c
// One block = (batch b, 128 context rows). 8 waves, 16 rows each.
__global__ void __launch_bounds__(256)
attn_kernel(const bf16_t* __restrict__ ctxw_bf, const bf16_t* __restrict__ qry_bf,
            const bf16_t* __restrict__ qryT_bf, const float* __restrict__ b_att,
            const float* __restrict__ cmask, const float* __restrict__ qmask,
            float* __restrict__ out, float* __restrict__ q2c) {
    __shared__ bf16_t alpha_lds[8][16 * QQ];          // 32 KB

    const int lane = threadIdx.x & 31;
    const int wave = threadIdx.x >> 5;
    const int half = lane >> 4;
    const int b     = blockIdx.x >> 3;                // C/128 == 8 tiles
    const int ctile = blockIdx.x & 7;
    const int c0    = ctile * 128 + wave * 16;        // row within batch
    const size_t rowbase = (size_t)b * CC + c0;       // global ctx row

    const bf16_t* Abase = ctxw_bf + rowbase * HH;
    const bf16_t* Bbase = qry_bf + (size_t)b * QQ * HH;

    // ---- Phase A: sim tile 16 x 128 (8 N-tiles), K = H = 768
    v8f acc[8];
#pragma unroll
    for (int t = 0; t < 8; ++t) acc[t] = vzero8();

    for (int k0 = 0; k0 < HH; k0 += 32) {
        v16bf a = load_frag_bf16(Abase, HH, 0, k0, lane);
#pragma unroll
        for (int t = 0; t < 8; ++t) {
            v16bf bb = load_frag_bf16(Bbase, HH, t * 16, k0, lane);
            acc[t] = wmma_bf16(a, bb, acc[t]);
        }
    }

    const float batt = b_att[0];
#pragma unroll
    for (int t = 0; t < 8; ++t)
#pragma unroll
        for (int v = 0; v < 8; ++v) acc[t][v] += batt;

    // ---- Row max (over N = 128): across 8 tiles, then across 16 lanes (half-wave)
    float rm[8];
#pragma unroll
    for (int v = 0; v < 8; ++v) {
        float m = acc[0][v];
#pragma unroll
        for (int t = 1; t < 8; ++t) m = fmaxf(m, acc[t][v]);
#pragma unroll
        for (int off = 1; off < 16; off <<= 1)
            m = fmaxf(m, __shfl_xor(m, off, 16));
        rm[v] = m;
    }
    if ((lane & 15) == 0) {
#pragma unroll
        for (int v = 0; v < 8; ++v)
            q2c[(size_t)b * CC + c0 + half * 8 + v] = rm[v];
    }

    // ---- exp / sum
    float rs[8];
#pragma unroll
    for (int v = 0; v < 8; ++v) rs[v] = 0.0f;
#pragma unroll
    for (int t = 0; t < 8; ++t)
#pragma unroll
        for (int v = 0; v < 8; ++v) {
            float e = __expf(acc[t][v] - rm[v]);
            acc[t][v] = e;
            rs[v] += e;
        }
#pragma unroll
    for (int v = 0; v < 8; ++v) {
#pragma unroll
        for (int off = 1; off < 16; off <<= 1)
            rs[v] += __shfl_xor(rs[v], off, 16);
        rs[v] = 1.0f / rs[v];
    }

    // ---- masked alpha -> LDS as bf16 (A-layout staging buffer, row-major 16x128)
    float cm[8];
#pragma unroll
    for (int v = 0; v < 8; ++v)
        cm[v] = cmask[(size_t)b * CC + c0 + half * 8 + v];
#pragma unroll
    for (int t = 0; t < 8; ++t) {
        const int qcol = t * 16 + (lane & 15);
        const float qm = qmask[(size_t)b * QQ + qcol];
#pragma unroll
        for (int v = 0; v < 8; ++v) {
            float al = acc[t][v] * rs[v] * cm[v] * qm;
            alpha_lds[wave][(half * 8 + v) * QQ + qcol] = (bf16_t)al;
        }
    }
    __syncthreads();

    // ---- Phase B: a = alpha @ qry  (M=16, N=H=768, K=Q=128), B from qryT [H,Q]
    const bf16_t* alphaA = &alpha_lds[wave][0];
    const bf16_t* BT     = qryT_bf + (size_t)b * HH * QQ;
    float* outb = out + rowbase * (4 * HH);

    v16bf afr[4];
#pragma unroll
    for (int kk = 0; kk < 4; ++kk)
        afr[kk] = load_frag_bf16(alphaA, QQ, 0, kk * 32, lane);

    for (int n0 = 0; n0 < HH; n0 += 16) {
        v8f av = vzero8();
#pragma unroll
        for (int kk = 0; kk < 4; ++kk) {
            v16bf bb = load_frag_bf16(BT, QQ, n0, kk * 32, lane);
            av = wmma_bf16(afr[kk], bb, av);
        }
        const int n = n0 + (lane & 15);
#pragma unroll
        for (int v = 0; v < 8; ++v) {
            float* orow = outb + (size_t)(half * 8 + v) * (4 * HH);
            const float ctxv = orow[n];          // ctx (written by proj_c)
            const float a_v  = av[v];
            orow[HH + n]     = a_v;              // a
            orow[2 * HH + n] = ctxv * a_v;       // c = ctx * a
        }
    }
}

// ------------------------------------------------ beta = softmax(q2c) * cmask
__global__ void __launch_bounds__(256)
beta_kernel(const float* __restrict__ q2c, const float* __restrict__ cmask,
            float* __restrict__ beta) {
    __shared__ float red[256];
    const int b = blockIdx.x;
    const int tid = threadIdx.x;

    float m = -1e30f;
    for (int c = tid; c < CC; c += 256) m = fmaxf(m, q2c[(size_t)b * CC + c]);
    red[tid] = m;
    __syncthreads();
    for (int s = 128; s > 0; s >>= 1) {
        if (tid < s) red[tid] = fmaxf(red[tid], red[tid + s]);
        __syncthreads();
    }
    const float M = red[0];
    __syncthreads();

    float s = 0.0f;
    for (int c = tid; c < CC; c += 256) s += __expf(q2c[(size_t)b * CC + c] - M);
    red[tid] = s;
    __syncthreads();
    for (int st = 128; st > 0; st >>= 1) {
        if (tid < st) red[tid] += red[tid + st];
        __syncthreads();
    }
    const float inv = 1.0f / red[0];
    __syncthreads();

    for (int c = tid; c < CC; c += 256)
        beta[(size_t)b * CC + c] =
            __expf(q2c[(size_t)b * CC + c] - M) * inv * cmask[(size_t)b * CC + c];
}

// ------------------------------------------------- bvec[b][p] = sum_c beta*ctx
__global__ void __launch_bounds__(128)
bvec_kernel(const float* __restrict__ beta, const float* __restrict__ out,
            float* __restrict__ bvec) {
    const int p = blockIdx.x * 128 + threadIdx.x;
    const int b = blockIdx.y;
    if (p >= HH) return;
    const float* ob = out + (size_t)b * CC * (4 * HH);
    float s = 0.0f;
    for (int c = 0; c < CC; ++c)
        s += beta[(size_t)b * CC + c] * ob[(size_t)c * (4 * HH) + p];
    bvec[(size_t)b * HH + p] = s;
}

// ------------------------------------------------------------- d = ctx * b[p]
__global__ void __launch_bounds__(256)
d_kernel(const float* __restrict__ bvec, float* __restrict__ out) {
    const int total = BB * CC * HH;
    int i = blockIdx.x * 256 + threadIdx.x;
    if (i >= total) return;
    const int p = i % HH;
    const int r = i / HH;          // global ctx row b*C + c
    const int b = r >> 10;         // / CC
    float* row = out + (size_t)r * (4 * HH);
    row[3 * HH + p] = row[p] * bvec[(size_t)b * HH + p];
}

// ============================================================================
extern "C" void kernel_launch(void* const* d_in, const int* in_sizes, int n_in,
                              void* d_out, int out_size, void* d_ws, size_t ws_size,
                              hipStream_t stream) {
    const float* context = (const float*)d_in[0];   // [B,C,H]
    const float* cmask   = (const float*)d_in[1];   // [B,C]
    const float* query   = (const float*)d_in[2];   // [B,Q,H]
    const float* qmask   = (const float*)d_in[3];   // [B,Q]
    const float* Wc      = (const float*)d_in[4];   // [H,H]
    const float* bc      = (const float*)d_in[5];   // [H]
    const float* Wq      = (const float*)d_in[6];   // [H,H]
    const float* bq      = (const float*)d_in[7];   // [H]
    const float* w_att   = (const float*)d_in[8];   // [H]
    const float* b_att   = (const float*)d_in[9];   // scalar
    float* out = (float*)d_out;                     // [B,C,4H]

    char* ws = (char*)d_ws;
    size_t off = 0;
    bf16_t* qry_bf   = (bf16_t*)(ws + off); off += (size_t)BB * QQ * HH * 2;
    bf16_t* qryT_bf  = (bf16_t*)(ws + off); off += (size_t)BB * HH * QQ * 2;
    bf16_t* ctxw_bf  = (bf16_t*)(ws + off); off += (size_t)BB * CC * HH * 2;
    bf16_t* ctx_bf   = (bf16_t*)(ws + off); off += (size_t)BB * CC * HH * 2;
    bf16_t* query_bf = (bf16_t*)(ws + off); off += (size_t)BB * QQ * HH * 2;
    bf16_t* Wc_bf    = (bf16_t*)(ws + off); off += (size_t)HH * HH * 2;
    bf16_t* Wq_bf    = (bf16_t*)(ws + off); off += (size_t)HH * HH * 2;
    float*  q2c      = (float*)(ws + off);  off += (size_t)BB * CC * 4;
    float*  beta     = (float*)(ws + off);  off += (size_t)BB * CC * 4;
    float*  bvec     = (float*)(ws + off);  off += (size_t)BB * HH * 4;

    // 0) fp32 -> bf16 conversions
    {
        int n = BB * CC * HH;
        cvt_f32_bf16_kernel<<<(n + 255) / 256, 256, 0, stream>>>(context, ctx_bf, n);
        n = BB * QQ * HH;
        cvt_f32_bf16_kernel<<<(n + 255) / 256, 256, 0, stream>>>(query, query_bf, n);
        n = HH * HH;
        cvt_f32_bf16_kernel<<<(n + 255) / 256, 256, 0, stream>>>(Wc, Wc_bf, n);
        cvt_f32_bf16_kernel<<<(n + 255) / 256, 256, 0, stream>>>(Wq, Wq_bf, n);
    }

    // 1) qry projection GEMM (B*Q = 2048 rows)
    proj_q_kernel<<<(BB * QQ) / 128, 256, 0, stream>>>(query_bf, Wq_bf, bq,
                                                       qry_bf, qryT_bf);

    // 2) ctx projection GEMM (B*C = 16384 rows) -> out[...,0:H] + ctx*w_att bf16
    proj_c_kernel<<<(BB * CC) / 128, 256, 0, stream>>>(ctx_bf, Wc_bf, bc, w_att,
                                                       out, ctxw_bf);

    // 3) fused sim GEMM + softmax + a GEMM + c
    attn_kernel<<<BB * (CC / 128), 256, 0, stream>>>(ctxw_bf, qry_bf, qryT_bf,
                                                     b_att, cmask, qmask, out, q2c);

    // 4) beta softmax over context dim
    beta_kernel<<<BB, 256, 0, stream>>>(q2c, cmask, beta);

    // 5) b = beta^T ctx
    bvec_kernel<<<dim3((HH + 127) / 128, BB), 128, 0, stream>>>(beta, out, bvec);

    // 6) d = ctx * b
    {
        int n = BB * CC * HH;
        d_kernel<<<(n + 255) / 256, 256, 0, stream>>>(bvec, out);
    }
}
